// Performer_46866683134729
// MI455X (gfx1250) — matrix operand order repaired and checked by reading, not compiled, not run on hardware
//
#include <hip/hip_runtime.h>

typedef __attribute__((ext_vector_type(16))) _Float16 v16h;
typedef __attribute__((ext_vector_type(8)))  float    v8f;
typedef __attribute__((ext_vector_type(8)))  _Float16 h8;
typedef __attribute__((ext_vector_type(4)))  float    f4;
typedef __attribute__((ext_vector_type(4)))  int      v4i;

#define B_   4
#define S_   4096
#define DM_  1024
#define H_   16
#define DH_  64
#define R_   256
#define BS_  (B_*S_)        // 16384 rows for the 1024x1024 GEMMs
#define BSH_ (B_*S_*H_)     // 262144 rows for the feature GEMMs

// ---------------------------------------------------------------------------
// gfx1250 async global->LDS copy (ASYNCcnt path).
// ---------------------------------------------------------------------------
#if defined(__HIP_DEVICE_COMPILE__) && __has_builtin(__builtin_amdgcn_global_load_async_to_lds_b128)
#define HAS_ASYNC 1
#else
#define HAS_ASYNC 0
#endif

#if HAS_ASYNC
typedef __attribute__((address_space(1))) v4i as1_v4i;
typedef __attribute__((address_space(3))) v4i as3_v4i;
__device__ __forceinline__ void cp16_async(_Float16* lds, const _Float16* g) {
  __builtin_amdgcn_global_load_async_to_lds_b128((as1_v4i*)g, (as3_v4i*)lds, 0, 0);
}
__device__ __forceinline__ void async_wait() {
#if __has_builtin(__builtin_amdgcn_s_wait_asynccnt)
  __builtin_amdgcn_s_wait_asynccnt(0);
#else
  asm volatile("s_wait_asynccnt 0x0" ::: "memory");
#endif
}
#else
__device__ __forceinline__ void cp16_async(_Float16* lds, const _Float16* g) {
  *(h8*)lds = *(const h8*)g;
}
__device__ __forceinline__ void async_wait() {}
#endif

// ---------------------------------------------------------------------------
// gfx1250 Tensor Data Mover: one 2D f16 tile global->LDS per call (TENSORcnt).
// D# layout per CDNA5 ISA. LDS rows: tile_d0*2 bytes of
// data then 16B padding (pad_interval=16 DWORDs -> code 3, pad_amount=4 DWORDs
// -> code 3), giving an 80B LDS pitch for 64B rows. This toolchain's builtin
// is the 6-arg form: (u32x4 g0, i32x8 g1, i32x4 g2, i32x4 g3, i32x8, i32 cpol).
// ---------------------------------------------------------------------------
#if defined(__HIP_DEVICE_COMPILE__) && __has_builtin(__builtin_amdgcn_tensor_load_to_lds) && __has_builtin(__builtin_amdgcn_s_wait_tensorcnt)
#define HAS_TDM 1
#else
#define HAS_TDM 0
#endif

#if HAS_TDM
typedef __attribute__((ext_vector_type(4))) unsigned int u32x4;
typedef __attribute__((ext_vector_type(8))) int i32x8;
typedef __attribute__((ext_vector_type(4))) int i32x4;
typedef __attribute__((address_space(3))) _Float16 as3_h;
__device__ __forceinline__ void tdm_load_tile_f16(
    _Float16* lds, const _Float16* g, int tile_d0, int tile_d1, int gstride) {
  unsigned lds_addr = (unsigned)(uintptr_t)(as3_h*)lds;
  unsigned long long ga = (unsigned long long)(uintptr_t)g;
  // group0: count=1 | lds_addr | global_addr[56:0] | type=2
  u32x4 g0 = { 1u, lds_addr, (unsigned)ga,
               (unsigned)((ga >> 32) & 0x01FFFFFFull) | (2u << 30) };
  // group1: data_size=2B, pad_enable, pad_interval=16dw, pad_amount=4dw,
  // tensor dims large (no OOB), tile dims, dim0 stride (data_size units)
  unsigned w0 = (1u << 16) | (1u << 20) | (3u << 22) | (3u << 25);
  unsigned dim0 = 0x40000000u, dim1 = 0x40000000u;
  i32x8 g1 = { (int)w0,
               (int)((dim0 & 0xFFFFu) << 16),
               (int)((dim0 >> 16) | ((dim1 & 0xFFFFu) << 16)),
               (int)((dim1 >> 16) | ((unsigned)tile_d0 << 16)),
               (int)(unsigned)tile_d1,
               (int)(unsigned)gstride,
               0, 0 };
  i32x4 z = { 0, 0, 0, 0 };
  i32x8 z8 = { 0, 0, 0, 0, 0, 0, 0, 0 };
  __builtin_amdgcn_tensor_load_to_lds(g0, g1, z, z, z8, 0);
}
__device__ __forceinline__ void tdm_wait() { __builtin_amdgcn_s_wait_tensorcnt(0); }
#endif

// ---------------------------------------------------------------------------
// WMMA fragment helpers: 16x16x32 f16, contiguous k = 16*(lane/16) + i, same order for A and B
// ---------------------------------------------------------------------------
__device__ __forceinline__ v16h frag_a16(const _Float16* base, int ld, int lane, int koff) {
  int m = lane & 15, h = lane >> 4;
  const _Float16* row = base + m * ld + koff + h * 16;
  v16h f;
#pragma unroll
  for (int i = 0; i < 16; ++i) f[i] = row[i];
  return f;
}

__device__ __forceinline__ v16h frag_bt16(const _Float16* base, int ld, int lane, int koff) {
  int n = lane & 15, h = lane >> 4;
  const _Float16* col = base + n * ld + koff + h * 16;
  v16h f;
#pragma unroll
  for (int i = 0; i < 16; ++i) f[i] = col[i];
  return f;
}

__device__ __forceinline__ v8f wmma32(v16h a, v16h b, v8f c) {
  return __builtin_amdgcn_wmma_f32_16x16x32_f16(false, a, false, b, (short)0, c, false, false);
}

// Ordered-uint encoding for float atomic max (monotone wrt float ordering)
__device__ __forceinline__ unsigned f2ord(float f) {
  unsigned u = __float_as_uint(f);
  return (u & 0x80000000u) ? ~u : (u | 0x80000000u);
}
__device__ __forceinline__ float ord2f(unsigned o) {
  return (o & 0x80000000u) ? __uint_as_float(o & 0x7FFFFFFFu) : __uint_as_float(~o);
}

__global__ void init_kernel(unsigned* gmax) { *gmax = f2ord(-3.0e38f); }

// ---------------------------------------------------------------------------
// Generic tiled GEMM with bias: C[M,N] = A[M,K] * W[K,N] + bias[N]
// Block tile 128x64, 8 waves. A=f16 stages via async global->LDS DMA.
// ---------------------------------------------------------------------------
template <typename AT, typename OT>
__global__ void __launch_bounds__(256) gemm_bias_kernel(
    const AT* __restrict__ A, const float* __restrict__ W,
    const float* __restrict__ bias, OT* __restrict__ C,
    int M, int N, int K) {
  const int LDA = 40, LDB = 40;  // halves; 80B rows -> 16B aligned chunks
  __shared__ __attribute__((aligned(16))) _Float16 sa[128 * 40];
  __shared__ __attribute__((aligned(16))) _Float16 sbt[64 * 40];
  int tid = threadIdx.x, lane = tid & 31, wave = tid >> 5;
  int bm = blockIdx.x * 128, bn = blockIdx.y * 64;
  int wn = (wave & 3) * 16, wm = (wave >> 2) * 64;
  v8f acc[4] = {};
  for (int k0 = 0; k0 < K; k0 += 32) {
    __syncthreads();
    if constexpr (sizeof(AT) == 2) {
      // A tile 128x32 halves = 512 x 16B chunks: async DMA, no VGPR bounce
#pragma unroll
      for (int j = 0; j < 2; ++j) {
        int idx = tid + j * 256;
        int r = idx >> 2, ch = (idx & 3) * 8;
        cp16_async(&sa[r * LDA + ch], (const _Float16*)&A[(size_t)(bm + r) * K + k0 + ch]);
      }
    } else {
      // A tile f32: 4x f4 vector loads + convert + 2x b128 LDS stores
      int r = tid >> 1, c0 = (tid & 1) * 16;
      const f4* gp = (const f4*)&A[(size_t)(bm + r) * K + k0 + c0];
      f4 x0 = gp[0], x1 = gp[1], x2 = gp[2], x3 = gp[3];
      h8 lo, hi;
#pragma unroll
      for (int j = 0; j < 4; ++j) {
        lo[j] = (_Float16)x0[j]; lo[4 + j] = (_Float16)x1[j];
        hi[j] = (_Float16)x2[j]; hi[4 + j] = (_Float16)x3[j];
      }
      *(h8*)&sa[r * LDA + c0] = lo;
      *(h8*)&sa[r * LDA + c0 + 8] = hi;
    }
    {
      // W tile 32x64 -> Bt[n][k]: 8 coalesced f32 loads, one b128 LDS store
      int n = tid & 63, kc = (tid >> 6) * 8;
      float wb[8];
#pragma unroll
      for (int j = 0; j < 8; ++j) wb[j] = W[(size_t)(k0 + kc + j) * N + bn + n];
      h8 wh;
#pragma unroll
      for (int j = 0; j < 8; ++j) wh[j] = (_Float16)wb[j];
      *(h8*)&sbt[n * LDB + kc] = wh;
    }
    if constexpr (sizeof(AT) == 2) async_wait();
    if (k0 + 32 < K)
      __builtin_prefetch(&A[(size_t)(bm + (tid >> 1)) * K + k0 + 32], 0, 1);
    __syncthreads();
    v16h b = frag_bt16(&sbt[wn * LDB], LDB, lane, 0);
#pragma unroll
    for (int t = 0; t < 4; ++t) {
      v16h a = frag_a16(&sa[(wm + t * 16) * LDA], LDA, lane, 0);
      acc[t] = wmma32(a, b, acc[t]);
    }
  }
  int n = bn + wn + (lane & 15);
  float bv = bias[n];
  int half = lane >> 4;
#pragma unroll
  for (int t = 0; t < 4; ++t)
#pragma unroll
    for (int r = 0; r < 8; ++r) {
      int m = bm + wm + t * 16 + half * 8 + r;
      C[(size_t)m * N + n] = (OT)(acc[t][r] + bv);
    }
}

// ---------------------------------------------------------------------------
// FAVOR+ feature maps: data_hat = (normalizer*X) @ proj^T, X:[BSH,64], proj:[256,64]
// MODE 0: global max of data_hat (keys). MODE 1: query (row max). MODE 2: key.
// ---------------------------------------------------------------------------
template <int MODE>
__global__ void __launch_bounds__(256) features_kernel(
    const _Float16* __restrict__ X, const float* __restrict__ proj,
    _Float16* __restrict__ Phi, unsigned* __restrict__ gmax) {
  const int LDA = 72, LDB = 72;  // 144B rows, 16B-aligned
  __shared__ __attribute__((aligned(16))) _Float16 sa[16 * 72];
  __shared__ __attribute__((aligned(16))) _Float16 sbt[256 * 72];
  __shared__ float sdiag[16];
  __shared__ float srow[8][16];
  __shared__ float sredm[8];
  const float normalizer = 0.3535533905932738f;  // 1/64^(1/4)
  const float ratio = 0.0625f;                   // 1/sqrt(256)
  int tid = threadIdx.x, lane = tid & 31, wave = tid >> 5;
  int m0 = blockIdx.x * 16;
  {
    int i = tid * 4, r = i >> 6, c = i & 63;
    const _Float16* gp = &X[(size_t)(m0 + r) * DH_ + c];
    _Float16* sp = &sa[r * LDA + c];
#pragma unroll
    for (int j = 0; j < 4; ++j) sp[j] = (_Float16)(normalizer * (float)gp[j]);
  }
  {
    const float* gp = &proj[tid * 64];
    _Float16* sp = &sbt[tid * LDB];
#pragma unroll
    for (int j = 0; j < 64; j += 8) {
      f4 a = *(const f4*)(gp + j), b = *(const f4*)(gp + j + 4);
      h8 hh;
#pragma unroll
      for (int q = 0; q < 4; ++q) { hh[q] = (_Float16)a[q]; hh[4 + q] = (_Float16)b[q]; }
      *(h8*)(sp + j) = hh;
    }
  }
  __syncthreads();
  if (MODE != 0 && tid < 16) {
    float s = 0.f;
    for (int c = 0; c < 64; ++c) { float v = (float)sa[tid * LDA + c]; s += v * v; }
    sdiag[tid] = 0.5f * s;  // 0.5*||normalizer*x||^2
  }
  v8f acc[2] = {};
#pragma unroll
  for (int kk = 0; kk < 2; ++kk) {
    v16h a = frag_a16(sa, LDA, lane, kk * 32);
#pragma unroll
    for (int t = 0; t < 2; ++t) {
      v16h b = frag_bt16(&sbt[(wave * 2 + t) * 16 * LDB], LDB, lane, kk * 32);
      acc[t] = wmma32(a, b, acc[t]);
    }
  }
  int half = lane >> 4;
  if (MODE == 0) {
    float mx = -3.0e38f;
#pragma unroll
    for (int t = 0; t < 2; ++t)
#pragma unroll
      for (int r = 0; r < 8; ++r) mx = fmaxf(mx, acc[t][r]);
    for (int d = 1; d < 32; d <<= 1) mx = fmaxf(mx, __shfl_xor(mx, d, 32));
    if (lane == 0) sredm[wave] = mx;
    __syncthreads();
    if (tid == 0) {
      float m2 = sredm[0];
      for (int w = 1; w < 8; ++w) m2 = fmaxf(m2, sredm[w]);
      atomicMax(gmax, f2ord(m2));
    }
    return;
  }
  float msub[8];
  if (MODE == 1) {
    float rmax[8];
#pragma unroll
    for (int r = 0; r < 8; ++r) {
      float v = fmaxf(acc[0][r], acc[1][r]);
      for (int d = 1; d < 16; d <<= 1) v = fmaxf(v, __shfl_xor(v, d, 32));
      rmax[r] = v;
    }
    if ((lane & 15) == 0) {
#pragma unroll
      for (int r = 0; r < 8; ++r) srow[wave][half * 8 + r] = rmax[r];
    }
    __syncthreads();
    if (tid < 16) {
      float v = srow[0][tid];
      for (int w = 1; w < 8; ++w) v = fmaxf(v, srow[w][tid]);
      srow[0][tid] = v;
    }
    __syncthreads();
#pragma unroll
    for (int r = 0; r < 8; ++r) msub[r] = srow[0][half * 8 + r];
  } else {
    __syncthreads();  // sdiag visibility
    float gm = ord2f(*gmax);
#pragma unroll
    for (int r = 0; r < 8; ++r) msub[r] = gm;
  }
#pragma unroll
  for (int t = 0; t < 2; ++t) {
    int n = wave * 32 + t * 16 + (lane & 15);
#pragma unroll
    for (int r = 0; r < 8; ++r) {
      int mloc = half * 8 + r;
      float v = ratio * (__expf(acc[t][r] - sdiag[mloc] - msub[r]) + 1e-4f);
      Phi[(size_t)(m0 + mloc) * R_ + n] = (_Float16)v;
    }
  }
}

// ---------------------------------------------------------------------------
// k1[b,h,r] = sum_s k_hat[b,s,h,r]
// ---------------------------------------------------------------------------
__global__ void __launch_bounds__(256) k1_kernel(
    const _Float16* __restrict__ khat, float* __restrict__ k1) {
  int bh = blockIdx.x, b = bh / H_, h = bh % H_;
  int r = threadIdx.x;
  float s = 0.f;
  for (int ss = 0; ss < S_; ++ss)
    s += (float)khat[((size_t)(b * S_ + ss) * H_ + h) * R_ + r];
  k1[(size_t)bh * R_ + r] = s;
}

// ---------------------------------------------------------------------------
// kv[b,h,r,v] = sum_s k_hat[b,s,h,r] * vfeat[b,s,h,v]   (per (b,h): 256x64, K=4096)
// Both operands need an LDS transpose: b128 global load -> 8 strided ds stores.
// ---------------------------------------------------------------------------
__global__ void __launch_bounds__(256) kv_kernel(
    const _Float16* __restrict__ khat, const _Float16* __restrict__ vh,
    float* __restrict__ kv) {
  const int LDA = 40, LDB = 40;
  __shared__ __attribute__((aligned(16))) _Float16 sa[64 * 40];
  __shared__ __attribute__((aligned(16))) _Float16 sbt[64 * 40];
  int tid = threadIdx.x, lane = tid & 31, wave = tid >> 5;
  int bh = blockIdx.x, b = bh / H_, h = bh % H_;
  int r0 = blockIdx.y * 64;
  v8f acc[2] = {};
  for (int s0 = 0; s0 < S_; s0 += 32) {
    __syncthreads();
    {
      int s = tid >> 3, rc = (tid & 7) * 8;
      h8 av = *(const h8*)&khat[((size_t)(b * S_ + s0 + s) * H_ + h) * R_ + r0 + rc];
      h8 bv = *(const h8*)&vh[((size_t)(b * S_ + s0 + s) * H_ + h) * DH_ + rc];
#pragma unroll
      for (int j = 0; j < 8; ++j) sa[(rc + j) * LDA + s] = av[j];
#pragma unroll
      for (int j = 0; j < 8; ++j) sbt[(rc + j) * LDB + s] = bv[j];
    }
    __syncthreads();
#pragma unroll
    for (int t = 0; t < 2; ++t) {
      int tt = wave * 2 + t, tm = tt >> 2, tn = tt & 3;
      v16h a = frag_a16(&sa[tm * 16 * LDA], LDA, lane, 0);
      v16h bb = frag_bt16(&sbt[tn * 16 * LDB], LDB, lane, 0);
      acc[t] = wmma32(a, bb, acc[t]);
    }
  }
  int half = lane >> 4;
#pragma unroll
  for (int t = 0; t < 2; ++t) {
    int tt = wave * 2 + t, tm = tt >> 2, tn = tt & 3;
    int n = tn * 16 + (lane & 15);
#pragma unroll
    for (int r = 0; r < 8; ++r) {
      int m = r0 + tm * 16 + half * 8 + r;
      kv[((size_t)bh * R_ + m) * DH_ + n] = acc[t][r];
    }
  }
}

// ---------------------------------------------------------------------------
// out[b,s,h,v] = (q_hat[b,s,h,:] @ kv[b,h,:,v]) / (q_hat[b,s,h,:] . k1[b,h,:])
// q_hat tile staged by the Tensor Data Mover: one tensor_load_to_lds per
// workgroup per k-step (wave 0 issues; TDM ignores EXEC), TENSORcnt waited
// before the barrier. Falls back to ASYNCcnt copies if TDM builtins absent.
// ---------------------------------------------------------------------------
__global__ void __launch_bounds__(256) attn_out_kernel(
    const _Float16* __restrict__ qhat, const float* __restrict__ kv,
    const float* __restrict__ k1, _Float16* __restrict__ outu) {
  const int LDA = 40, LDB = 40;
  __shared__ __attribute__((aligned(16))) _Float16 sa[64 * 40];
  __shared__ __attribute__((aligned(16))) _Float16 sbt[64 * 40];
  __shared__ float sden[64];
  int tid = threadIdx.x, lane = tid & 31, wave = tid >> 5;
  int bh = blockIdx.x, b = bh / H_, h = bh % H_;
  int m0 = blockIdx.y * 64;  // s offset
  if (tid < 64) {
    float s = 0.f;
    const _Float16* row = &qhat[((size_t)(b * S_ + m0 + tid) * H_ + h) * R_];
    const float* kk = &k1[(size_t)bh * R_];
    for (int r = 0; r < R_; ++r) s += (float)row[r] * kk[r];
    sden[tid] = s;
  }
  v8f acc[2] = {};
  for (int r0 = 0; r0 < R_; r0 += 32) {
    __syncthreads();
#if HAS_TDM
    if (tid < 32) {
      // A: q_hat 2D tile, 64 rows x 32 halves, row stride H*R elements,
      // DMA'd by the TDM with 64B->80B LDS row padding.
      tdm_load_tile_f16(sa, &qhat[((size_t)(b * S_ + m0) * H_ + h) * R_ + r0],
                        32, 64, H_ * R_);
    }
#else
    {
      int s = tid >> 2, ch = (tid & 3) * 8;
      cp16_async(&sa[s * LDA + ch],
                 &qhat[((size_t)(b * S_ + m0 + s) * H_ + h) * R_ + r0 + ch]);
    }
#endif
    {
      // B: kv f32 transpose -> Bt[v][r]: 8 coalesced loads + one b128 store
      int v = tid & 63, rc = (tid >> 6) * 8;
      float kb[8];
#pragma unroll
      for (int j = 0; j < 8; ++j) kb[j] = kv[((size_t)bh * R_ + r0 + rc + j) * DH_ + v];
      h8 kh8;
#pragma unroll
      for (int j = 0; j < 8; ++j) kh8[j] = (_Float16)kb[j];
      *(h8*)&sbt[v * LDB + rc] = kh8;
    }
#if HAS_TDM
    if (tid < 32) tdm_wait();
#else
    async_wait();
#endif
    __syncthreads();
#pragma unroll
    for (int t = 0; t < 2; ++t) {
      int tt = wave * 2 + t, tm = tt >> 2, tn = tt & 3;
      v16h a = frag_a16(&sa[tm * 16 * LDA], LDA, lane, 0);
      v16h bb = frag_bt16(&sbt[tn * 16 * LDB], LDB, lane, 0);
      acc[t] = wmma32(a, bb, acc[t]);
    }
  }
  int half = lane >> 4;
#pragma unroll
  for (int t = 0; t < 2; ++t) {
    int tt = wave * 2 + t, tm = tt >> 2, tn = tt & 3;
    int n = tn * 16 + (lane & 15);
#pragma unroll
    for (int r = 0; r < 8; ++r) {
      int ms = tm * 16 + half * 8 + r;
      float v = acc[t][r] / sden[ms];
      outu[((size_t)(b * S_ + m0 + ms) * H_ + h) * DH_ + n] = (_Float16)v;
    }
  }
}

// ---------------------------------------------------------------------------
extern "C" void kernel_launch(void* const* d_in, const int* in_sizes, int n_in,
                              void* d_out, int out_size, void* d_ws, size_t ws_size,
                              hipStream_t stream) {
  (void)in_sizes; (void)n_in; (void)out_size; (void)ws_size;
  const float* query = (const float*)d_in[0];
  const float* key   = (const float*)d_in[1];
  const float* value = (const float*)d_in[2];
  const float* Wq = (const float*)d_in[3];
  const float* bq = (const float*)d_in[4];
  const float* Wk = (const float*)d_in[5];
  const float* bk = (const float*)d_in[6];
  const float* Wv = (const float*)d_in[7];
  const float* bv = (const float*)d_in[8];
  const float* Wo = (const float*)d_in[9];
  const float* bo = (const float*)d_in[10];
  const float* proj = (const float*)d_in[11];

  char* ws = (char*)d_ws;
  size_t off = 0;
  auto carve = [&](size_t bytes) {
    char* p = ws + off;
    off += (bytes + 255) & ~(size_t)255;
    return p;
  };
  _Float16* qh   = (_Float16*)carve((size_t)BS_ * DM_ * 2);   // q  [B,S,H,Dh] f16
  _Float16* kh   = (_Float16*)carve((size_t)BS_ * DM_ * 2);   // k
  _Float16* vh   = (_Float16*)carve((size_t)BS_ * DM_ * 2);   // v
  _Float16* qhat = (_Float16*)carve((size_t)BSH_ * R_ * 2);   // q_hat [B,S,H,R]
  _Float16* khat = (_Float16*)carve((size_t)BSH_ * R_ * 2);   // k_hat
  float* kvbuf   = (float*)carve((size_t)B_ * H_ * R_ * DH_ * 4);
  float* k1buf   = (float*)carve((size_t)B_ * H_ * R_ * 4);
  _Float16* outu = (_Float16*)carve((size_t)BS_ * DM_ * 2);   // pre-proj out
  unsigned* gmax = (unsigned*)carve(256);

  dim3 blk(256);
  init_kernel<<<1, 1, 0, stream>>>(gmax);
  dim3 g1(BS_ / 128, DM_ / 64);
  gemm_bias_kernel<float, _Float16><<<g1, blk, 0, stream>>>(query, Wq, bq, qh, BS_, DM_, DM_);
  gemm_bias_kernel<float, _Float16><<<g1, blk, 0, stream>>>(key,   Wk, bk, kh, BS_, DM_, DM_);
  gemm_bias_kernel<float, _Float16><<<g1, blk, 0, stream>>>(value, Wv, bv, vh, BS_, DM_, DM_);
  dim3 gf(BSH_ / 16);
  features_kernel<0><<<gf, blk, 0, stream>>>(kh, proj, nullptr, gmax);  // global key max
  features_kernel<1><<<gf, blk, 0, stream>>>(qh, proj, qhat, gmax);     // query features
  features_kernel<2><<<gf, blk, 0, stream>>>(kh, proj, khat, gmax);     // key features
  k1_kernel<<<dim3(B_ * H_), blk, 0, stream>>>(khat, k1buf);
  kv_kernel<<<dim3(B_ * H_, R_ / 64), blk, 0, stream>>>(khat, vh, kvbuf);
  attn_out_kernel<<<dim3(B_ * H_, S_ / 64), blk, 0, stream>>>(qhat, kvbuf, k1buf, outu);
  gemm_bias_kernel<_Float16, float><<<g1, blk, 0, stream>>>(outu, Wo, bo, (float*)d_out, BS_, DM_, DM_);
}
